// gnn_lstm_diagnoser_62852551410034
// MI455X (gfx1250) — compile-verified, matrix operand order checked by reading, not run on hardware
//
#include <hip/hip_runtime.h>
#include <hip/hip_bf16.h>

typedef __bf16 bf16;
typedef __attribute__((ext_vector_type(16))) __bf16 v16bf;
typedef __attribute__((ext_vector_type(8)))  __bf16 v8bf;
typedef __attribute__((ext_vector_type(8)))  float  v8f;

#define N_NODES 5
#define H 64
#define TIME 1024
#define BATCH 256
#define GATES 256     // 4*H
#define K1 96         // layer-1 K: [x-pad(32) | h1(64)]
#define K2 128        // layer-2 K: [h1(64) | h2(64)]
#define UCOLS 320     // [x p0(32)|x p1(32)|h1 p0(64)|h1 p1(64)|h2 p0(64)|h2 p1(64)]
#define XOFF 0
#define H1OFF 64
#define H2OFF 192
#define FC 256
#define NCLS 7
#define LOG2E 1.4426950408889634f

__device__ __forceinline__ v16bf ldfrag(const bf16* p) {
  // 16x32 bf16 fragment: two 16B chunks 32B apart (wave32 A/B layout).
  union { v16bf v; v8bf h[2]; } u;
  u.h[0] = *(const v8bf*)(p);
  u.h[1] = *(const v8bf*)(p + 16);
  return u.v;
}

// Branch-free activations on native v_exp_f32 / v_rcp_f32 (no EXEC divergence
// inside the WMMA loop, unlike OCML tanhf).
__device__ __forceinline__ float fast_sigmoid(float x) {
  return __builtin_amdgcn_rcpf(1.0f + __builtin_amdgcn_exp2f(-x * LOG2E));
}
__device__ __forceinline__ float fast_tanh(float x) {
  // tanh(x) = 1 - 2/(exp(2x)+1); exp2 overflow/underflow saturates correctly.
  float ex = __builtin_amdgcn_exp2f(x * (2.0f * LOG2E));
  return 1.0f - 2.0f * __builtin_amdgcn_rcpf(ex + 1.0f);
}

// ---------------------------------------------------------------------------
// Pack x: xg[n][t][b][0..3] = { x[b,0,nbr0,t], x[b,0,nbr1,t], x[b,0,nbr2,t], 0 }
// ---------------------------------------------------------------------------
__global__ void pack_x_kernel(const float* __restrict__ x, const int* __restrict__ gidx,
                              bf16* __restrict__ xg) {
  int n = blockIdx.y;
  int t = blockIdx.x;
  int b = threadIdx.x;
  int i0 = gidx[n*3+0], i1 = gidx[n*3+1], i2 = gidx[n*3+2];
  const float* xb = x + (size_t)b * (N_NODES*TIME);
  float v0 = xb[(size_t)i0*TIME + t];
  float v1 = xb[(size_t)i1*TIME + t];
  float v2 = xb[(size_t)i2*TIME + t];
  bf16* dst = xg + (((size_t)n*TIME + t)*BATCH + b)*4;
  dst[0] = (bf16)v0; dst[1] = (bf16)v1; dst[2] = (bf16)v2; dst[3] = (bf16)0.0f;
}

// ---------------------------------------------------------------------------
// Pack combined weights (bf16) and summed biases (f32) per node.
// W1c[n][g][k] (K1=96): k<3 = w_ih0, 3..31 = 0, 32..95 = w_hh0[k-32]
// W2c[n][g][k] (K2=128): k<64 = w_ih1, 64..127 = w_hh1[k-64]
// ---------------------------------------------------------------------------
__global__ void pack_w_kernel(const float* __restrict__ wih0, const float* __restrict__ whh0,
                              const float* __restrict__ bih0, const float* __restrict__ bhh0,
                              const float* __restrict__ wih1, const float* __restrict__ whh1,
                              const float* __restrict__ bih1, const float* __restrict__ bhh1,
                              bf16* __restrict__ W1c, bf16* __restrict__ W2c,
                              float* __restrict__ bias1, float* __restrict__ bias2) {
  int n = blockIdx.x;
  int g = threadIdx.x;   // 0..255 gate row
  bf16* w1 = W1c + ((size_t)n*GATES + g)*K1;
  bf16* w2 = W2c + ((size_t)n*GATES + g)*K2;
  for (int k = 0; k < 32; ++k) {
    float a = (k < 3) ? wih0[((size_t)n*GATES + g)*3 + k] : 0.0f;
    w1[k] = (bf16)a;
  }
  for (int k = 0; k < 64; ++k) {
    w1[32 + k] = (bf16)whh0[((size_t)n*GATES + g)*64 + k];
    w2[k]      = (bf16)wih1[((size_t)n*GATES + g)*64 + k];
    w2[64 + k] = (bf16)whh1[((size_t)n*GATES + g)*64 + k];
  }
  bias1[n*GATES + g] = bih0[n*GATES + g] + bhh0[n*GATES + g];
  bias2[n*GATES + g] = bih1[n*GATES + g] + bhh1[n*GATES + g];
}

// ---------------------------------------------------------------------------
// Persistent 2-layer LSTM per node. 512 threads = 16 waves; wave = M-block
// (16 batch rows). Each wave owns its rows end-to-end: h1/h2 LDS round trips
// are intra-wave (C-layout -> A-layout shuffle), so only the x tile needs a
// workgroup barrier. x/h1/h2 column blocks are ping-ponged by time parity.
// 16 N-tiles processed as 4 passes of 4 gate-tiles -> peak 32 acc VGPRs.
// Accumulator chains start from a shared zero (folds to inline C=0); biases
// live in LDS and are added in the elementwise pass.
// ---------------------------------------------------------------------------
__global__ __launch_bounds__(512) void lstm_main_kernel(
    const bf16* __restrict__ xg, const bf16* __restrict__ W1c, const bf16* __restrict__ W2c,
    const float* __restrict__ bias1, const float* __restrict__ bias2,
    float* __restrict__ feat) {
  extern __shared__ __align__(16) char smem[];
  bf16*  U   = (bf16*)smem;               // BATCH x UCOLS      (160 KB)
  bf16*  W1s = U + BATCH*UCOLS;           // GATES x K1         (48 KB)
  bf16*  W2s = W1s + (size_t)GATES*K1;    // GATES x K2         (64 KB)
  float* Bs1 = (float*)(W2s + (size_t)GATES*K2);  // 256 f32 (1 KB)
  float* Bs2 = Bs1 + GATES;                        // 256 f32 (1 KB)
  const int n   = blockIdx.x;
  const int tid = threadIdx.x;

  // Stage weights + biases into LDS; zero-init U.
  {
    const uint4* s1 = (const uint4*)(W1c + (size_t)n*GATES*K1);
    const uint4* s2 = (const uint4*)(W2c + (size_t)n*GATES*K2);
    uint4* d1 = (uint4*)W1s;
    uint4* d2 = (uint4*)W2s;
    for (int i = tid; i < GATES*K1/8; i += 512) d1[i] = s1[i];   // 3072
    for (int i = tid; i < GATES*K2/8; i += 512) d2[i] = s2[i];   // 4096
    uint4 z; z.x = z.y = z.z = z.w = 0u;
    uint4* du = (uint4*)U;
    for (int i = tid; i < BATCH*UCOLS/8; i += 512) du[i] = z;    // 10240
    for (int i = tid; i < GATES; i += 512) {
      Bs1[i] = bias1[n*GATES + i];
      Bs2[i] = bias2[n*GATES + i];
    }
  }

  const int wave = tid >> 5;              // == mblk, 0..15
  const int lane = tid & 31;
  const int lrow = lane & 15;
  const int hi8  = (lane >= 16) ? 8 : 0;
  const int arow = (wave*16 + lrow) * UCOLS;

  // Cell state in registers: c[qi][e] for (hid = qi*16+lrow, row = wave*16+e+hi8).
  float c1[4][8], c2[4][8];
#pragma unroll
  for (int qi = 0; qi < 4; ++qi)
#pragma unroll
    for (int e = 0; e < 8; ++e) { c1[qi][e] = 0.f; c2[qi][e] = 0.f; }

  // Shared zero accumulator tuple: first WMMA of each chain consumes it
  // (compiler folds this to the inline C=0 operand).
  v8f zacc;
#pragma unroll
  for (int e = 0; e < 8; ++e) zacc[e] = 0.f;

  const uint2* xrows = (const uint2*)(xg + (size_t)n*TIME*BATCH*4);

  __syncthreads();

#pragma unroll 1
  for (int t = 0; t < TIME; ++t) {
    const int p  = t & 1;
    const int po = p ^ 1;
    const int xw  = XOFF  + 32*p;    // x block written+read this step
    const int h1w = H1OFF + 64*p;    // h1 written this step
    const int h1r = H1OFF + 64*po;   // h1 from step t-1
    const int h2w = H2OFF + 64*p;    // h2 written this step
    const int h2r = H2OFF + 64*po;   // h2 from step t-1

    // ---- stage x_t (one uint2 = 4 bf16 per batch row) ----
    if (tid < BATCH) {
      uint2 v = xrows[t*BATCH + tid];
      *(uint2*)(U + tid*UCOLS + xw) = v;
    }
    __syncthreads();   // the ONLY barrier per step: x visible to all waves

    // ================= layer 1: gates = [x|h1_old] @ W1c^T =================
#pragma unroll
    for (int qi = 0; qi < 4; ++qi) {
      v8f acc[4];
#pragma unroll
      for (int ks = 0; ks < 3; ++ks) {
        int colbase = (ks == 0) ? xw : (h1r + (ks-1)*32);
        v16bf af = ldfrag(U + arow + colbase + hi8);
#pragma unroll
        for (int g = 0; g < 4; ++g) {
          int tile = g*4 + qi;
          v16bf bfrag = ldfrag(W1s + (tile*16 + lrow)*K1 + ks*32 + hi8);
          acc[g] = __builtin_amdgcn_wmma_f32_16x16x32_bf16(
              false, af, false, bfrag, (short)0,
              (ks == 0) ? zacc : acc[g], false, false);
        }
      }
      // cell update for this qi; writes only this wave's own rows (no race).
      int hid = qi*16 + lrow;
      float bi0 = Bs1[hid], bi1 = Bs1[64 + hid], bi2 = Bs1[128 + hid], bi3 = Bs1[192 + hid];
#pragma unroll
      for (int e = 0; e < 8; ++e) {
        float iv = fast_sigmoid(acc[0][e] + bi0);
        float fv = fast_sigmoid(acc[1][e] + bi1);
        float gv = fast_tanh(acc[2][e] + bi2);
        float ov = fast_sigmoid(acc[3][e] + bi3);
        float c = fv*c1[qi][e] + iv*gv;
        c1[qi][e] = c;
        float h = ov * fast_tanh(c);
        int row = wave*16 + e + hi8;
        U[row*UCOLS + h1w + hid] = (bf16)h;
      }
    }

    // ================= layer 2: gates = [h1_new|h2_old] @ W2c^T =================
    // h1_new rows needed are this wave's own writes; DScnt orders them in-wave.
#pragma unroll
    for (int qi = 0; qi < 4; ++qi) {
      v8f acc[4];
#pragma unroll
      for (int ks = 0; ks < 4; ++ks) {
        int colbase = (ks < 2) ? (h1w + ks*32) : (h2r + (ks-2)*32);
        v16bf af = ldfrag(U + arow + colbase + hi8);
#pragma unroll
        for (int g = 0; g < 4; ++g) {
          int tile = g*4 + qi;
          v16bf bfrag = ldfrag(W2s + (tile*16 + lrow)*K2 + ks*32 + hi8);
          acc[g] = __builtin_amdgcn_wmma_f32_16x16x32_bf16(
              false, af, false, bfrag, (short)0,
              (ks == 0) ? zacc : acc[g], false, false);
        }
      }
      int hid = qi*16 + lrow;
      float bi0 = Bs2[hid], bi1 = Bs2[64 + hid], bi2 = Bs2[128 + hid], bi3 = Bs2[192 + hid];
#pragma unroll
      for (int e = 0; e < 8; ++e) {
        float iv = fast_sigmoid(acc[0][e] + bi0);
        float fv = fast_sigmoid(acc[1][e] + bi1);
        float gv = fast_tanh(acc[2][e] + bi2);
        float ov = fast_sigmoid(acc[3][e] + bi3);
        float c = fv*c2[qi][e] + iv*gv;
        c2[qi][e] = c;
        float h = ov * fast_tanh(c);
        int row = wave*16 + e + hi8;
        U[row*UCOLS + h2w + hid] = (bf16)h;
        if (t == TIME-1) {
          feat[(size_t)row*(N_NODES*H) + n*H + hid] = h;   // feat[b][n*64+hid]
        }
      }
    }
  }
}

// ---------------------------------------------------------------------------
// Head 1: y = relu(feat @ w1^T + b1); batchnorm over batch (population var).
// ---------------------------------------------------------------------------
__global__ void head1_kernel(const float* __restrict__ feat, const float* __restrict__ w1,
                             const float* __restrict__ b1, const float* __restrict__ gamma,
                             const float* __restrict__ beta,
                             float* __restrict__ ybuf, float* __restrict__ y2buf) {
  int f = threadIdx.x;            // 0..255
  const float* wrow = w1 + (size_t)f*(N_NODES*H);
  float sum = 0.f, ss = 0.f;
  for (int b = 0; b < BATCH; ++b) {
    const float* fr = feat + (size_t)b*(N_NODES*H);
    float d = b1[f];
    for (int k = 0; k < N_NODES*H; ++k) d += fr[k]*wrow[k];
    d = fmaxf(d, 0.f);
    ybuf[(size_t)b*FC + f] = d;
    sum += d; ss += d*d;
  }
  float mean = sum * (1.0f/BATCH);
  float var  = ss * (1.0f/BATCH) - mean*mean;
  float inv  = rsqrtf(var + 1e-5f);
  float ga = gamma[f], be = beta[f];
  for (int b = 0; b < BATCH; ++b)
    y2buf[(size_t)b*FC + f] = (ybuf[(size_t)b*FC + f] - mean)*inv*ga + be;
}

// ---------------------------------------------------------------------------
// Head 2: logits = y2 @ w2^T + b2; softmax per row.
// ---------------------------------------------------------------------------
__global__ void head2_kernel(const float* __restrict__ y2, const float* __restrict__ w2,
                             const float* __restrict__ b2, float* __restrict__ out) {
  int b = threadIdx.x;            // 0..255
  const float* yr = y2 + (size_t)b*FC;
  float lg[NCLS];
  float mx = -1e30f;
  for (int c = 0; c < NCLS; ++c) {
    const float* wr = w2 + (size_t)c*FC;
    float d = b2[c];
    for (int k = 0; k < FC; ++k) d += yr[k]*wr[k];
    lg[c] = d;
    mx = fmaxf(mx, d);
  }
  float s = 0.f;
  for (int c = 0; c < NCLS; ++c) {
    float e = __builtin_amdgcn_exp2f((lg[c]-mx) * LOG2E);
    lg[c] = e; s += e;
  }
  float inv = 1.0f/s;
  for (int c = 0; c < NCLS; ++c) out[(size_t)b*NCLS + c] = lg[c]*inv;
}

// ---------------------------------------------------------------------------
extern "C" void kernel_launch(void* const* d_in, const int* in_sizes, int n_in,
                              void* d_out, int out_size, void* d_ws, size_t ws_size,
                              hipStream_t stream) {
  (void)in_sizes; (void)n_in; (void)out_size; (void)ws_size;
  const float* x     = (const float*)d_in[0];
  const int*   gidx  = (const int*)  d_in[1];
  const float* wih0  = (const float*)d_in[2];
  const float* whh0  = (const float*)d_in[3];
  const float* bih0  = (const float*)d_in[4];
  const float* bhh0  = (const float*)d_in[5];
  const float* wih1  = (const float*)d_in[6];
  const float* whh1  = (const float*)d_in[7];
  const float* bih1  = (const float*)d_in[8];
  const float* bhh1  = (const float*)d_in[9];
  const float* w1    = (const float*)d_in[10];
  const float* b1    = (const float*)d_in[11];
  const float* gamma = (const float*)d_in[12];
  const float* beta  = (const float*)d_in[13];
  const float* w2    = (const float*)d_in[14];
  const float* b2    = (const float*)d_in[15];
  float* out = (float*)d_out;

  char* ws = (char*)d_ws;
  size_t off = 0;
  bf16*  xg    = (bf16*)(ws + off); off += (size_t)N_NODES*TIME*BATCH*4*sizeof(bf16);
  bf16*  W1c   = (bf16*)(ws + off); off += (size_t)N_NODES*GATES*K1*sizeof(bf16);
  bf16*  W2c   = (bf16*)(ws + off); off += (size_t)N_NODES*GATES*K2*sizeof(bf16);
  float* bias1 = (float*)(ws + off); off += (size_t)N_NODES*GATES*sizeof(float);
  float* bias2 = (float*)(ws + off); off += (size_t)N_NODES*GATES*sizeof(float);
  float* feat  = (float*)(ws + off); off += (size_t)BATCH*N_NODES*H*sizeof(float);
  float* ybuf  = (float*)(ws + off); off += (size_t)BATCH*FC*sizeof(float);
  float* y2buf = (float*)(ws + off); off += (size_t)BATCH*FC*sizeof(float);

  pack_x_kernel<<<dim3(TIME, N_NODES), BATCH, 0, stream>>>(x, gidx, xg);
  pack_w_kernel<<<N_NODES, GATES, 0, stream>>>(wih0, whh0, bih0, bhh0,
                                               wih1, whh1, bih1, bhh1,
                                               W1c, W2c, bias1, bias2);
  size_t smem = (size_t)(BATCH*UCOLS + GATES*K1 + GATES*K2) * sizeof(bf16)
              + (size_t)2*GATES*sizeof(float);   // 272 KB + 2 KB LDS
  lstm_main_kernel<<<N_NODES, 512, smem, stream>>>(xg, W1c, W2c, bias1, bias2, feat);
  head1_kernel<<<1, FC, 0, stream>>>(feat, w1, b1, gamma, beta, ybuf, y2buf);
  head2_kernel<<<1, BATCH, 0, stream>>>(y2buf, w2, b2, out);
}
